// Model_66340064854739
// MI455X (gfx1250) — matrix operand order repaired and checked by reading, not compiled, not run on hardware
//
#include <hip/hip_runtime.h>
#include <hip/hip_bf16.h>
#include <stdint.h>

typedef long long i64;
typedef __attribute__((ext_vector_type(16))) __bf16 v16bf;
typedef __attribute__((ext_vector_type(8)))  float  v8f;

union FragU { unsigned u[8]; v16bf v; };

__device__ __forceinline__ unsigned short f2bf(float f) {
    union { float f; unsigned u; } c; c.f = f;
    unsigned r = 0x7FFFu + ((c.u >> 16) & 1u);
    return (unsigned short)((c.u + r) >> 16);
}

// Fragment loaders per ISA bf16 VGPR layouts (A 16x32 / B 32x16), from LDS
// tiles stored as [row][32] bf16 row-major (B is staged transposed: [n][k]).
__device__ __forceinline__ FragU loadA(const unsigned short* As, int row, int lane) {
    int m = lane & 15, h = lane >> 4;
    FragU f;
    const unsigned* ap = (const unsigned*)(As + (row + m) * 32);
#pragma unroll
    for (int i = 0; i < 8; ++i) {
        int kb = 2 * i + ((i >= 4) ? 8 : 0) + 8 * h;
        f.u[i] = ap[kb >> 1];
    }
    return f;
}
__device__ __forceinline__ FragU loadB(const unsigned short* Bs, int row, int lane) {
    int n = lane & 15, h = lane >> 4;
    FragU f;
    const unsigned* bp = (const unsigned*)(Bs + (row + n) * 32);
#pragma unroll
    for (int i = 0; i < 8; ++i) {
        int k0 = 2 * i + ((i >= 4) ? 8 : 0) + 8 * h;
        f.u[i] = bp[k0 >> 1];
    }
    return f;
}

struct Acc4 { v8f a[2][2]; };

__device__ __forceinline__ void acc_zero(Acc4& A) {
#pragma unroll
    for (int i = 0; i < 2; ++i)
#pragma unroll
        for (int j = 0; j < 2; ++j)
            A.a[i][j] = (v8f){0.f, 0.f, 0.f, 0.f, 0.f, 0.f, 0.f, 0.f};
}

// One K-step for a 32x32 wave tile: 2 A-frags x 2 B-frags -> 4 WMMAs.
__device__ __forceinline__ void wmma_step(const unsigned short* As, const unsigned short* Bs,
                                          int aRow, int bRow, int lane, Acc4& acc) {
    FragU a0 = loadA(As, aRow,      lane);
    FragU a1 = loadA(As, aRow + 16, lane);
    FragU b0 = loadB(Bs, bRow,      lane);
    FragU b1 = loadB(Bs, bRow + 16, lane);
    acc.a[0][0] = __builtin_amdgcn_wmma_f32_16x16x32_bf16(false, a0.v, false, b0.v, (short)0, acc.a[0][0], false, false);
    acc.a[0][1] = __builtin_amdgcn_wmma_f32_16x16x32_bf16(false, a0.v, false, b1.v, (short)0, acc.a[0][1], false, false);
    acc.a[1][0] = __builtin_amdgcn_wmma_f32_16x16x32_bf16(false, a1.v, false, b0.v, (short)0, acc.a[1][0], false, false);
    acc.a[1][1] = __builtin_amdgcn_wmma_f32_16x16x32_bf16(false, a1.v, false, b1.v, (short)0, acc.a[1][1], false, false);
}

// ---------------------------------------------------------------------------
// Generic conv-as-GEMM: out[z][m][p] = epi( sum_k W[m][k] * im2col(X[z])[k][p] )
// p = t*150 + v (T=20, V=150, N=3000). K = Cin*KH*KW. KH/KW compile-time.
// Block: 128 thr = 4 waves (2x2), 64x64 output tile, 32x32 per wave.
// ---------------------------------------------------------------------------
template <int KH, int KW>
__global__ __launch_bounds__(128)
void k_gemm_conv(const float* __restrict__ X, const float* __restrict__ W,
                 const float* __restrict__ bias, const float* __restrict__ bnS,
                 const float* __restrict__ bnT, const float* __restrict__ res,
                 const float* __restrict__ pe, float* __restrict__ out,
                 int M, int K, int padH, int padW,
                 i64 xStride, i64 outStride, i64 resStride, int doLrelu) {
    constexpr int Tdim = 20, Vdim = 150, N = 3000;
    constexpr int khkw = KH * KW;
    __shared__ unsigned short As[64 * 32];
    __shared__ unsigned short Bs[64 * 32];
    int z = blockIdx.z;
    const float* Xz = X + (i64)z * xStride;
    int tM = blockIdx.y * 64, tN = blockIdx.x * 64;
    int tid = threadIdx.x, lane = tid & 31, wave = tid >> 5;
    int wm = wave >> 1, wn = wave & 1;
    Acc4 acc; acc_zero(acc);

    for (int k0 = 0; k0 < K; k0 += 32) {
#pragma unroll
        for (int e = tid; e < 2048; e += 128) {
            int r = e >> 5, c = e & 31;
            int kg = k0 + c;
            int mg = tM + r;
            float av = (mg < M && kg < K) ? W[(i64)mg * K + kg] : 0.f;
            As[e] = f2bf(av);
            float bv = 0.f;
            int ng = tN + r;
            if (ng < N && kg < K) {
                int ic = kg / khkw; int rr = kg - ic * khkw;   // constant divisor
                int kh = rr / KW;  int kw = rr - kh * KW;      // constant divisor
                int t = ng / Vdim, v = ng - t * Vdim;          // constant divisor
                int tt = t + kh - padH, vv = v + kw - padW;
                if (tt >= 0 && tt < Tdim && vv >= 0 && vv < Vdim) {
                    bv = Xz[(i64)ic * (Tdim * Vdim) + tt * Vdim + vv];
                    if (pe) bv += pe[ic * Vdim + vv];
                }
            }
            Bs[e] = f2bf(bv);
        }
        __syncthreads();
        wmma_step(As, Bs, wm * 32, wn * 32, lane, acc);
        __syncthreads();
    }

    int nLoc = lane & 15, mOff = (lane >> 4) * 8;
    float* outz = out + (i64)z * outStride;
    const float* resz = res ? res + (i64)z * resStride : (const float*)0;
#pragma unroll
    for (int i = 0; i < 2; ++i)
#pragma unroll
        for (int j = 0; j < 2; ++j) {
            int ng = tN + wn * 32 + j * 16 + nLoc;
#pragma unroll
            for (int r = 0; r < 8; ++r) {
                int mg = tM + wm * 32 + i * 16 + mOff + r;
                if (mg < M && ng < N) {
                    float y = acc.a[i][j][r];
                    if (bias) y += bias[mg];
                    if (bnS) y = y * bnS[mg] + bnT[mg];
                    if (resz) y += resz[(i64)mg * N + ng];
                    if (doLrelu) y = (y >= 0.f) ? y : 0.1f * y;
                    outz[(i64)mg * N + ng] = y;
                }
            }
        }
}

// ---------------------------------------------------------------------------
// Attention scores: att[n,h,u,v] = tanh((q^T k)/(d*T))*alpha[h]*aw[u,v]+att0[h,u,v]
// q,k rows (c*20+t) x 150 of qkv output; K = d*20.
// ---------------------------------------------------------------------------
__global__ __launch_bounds__(128)
void k_attn(const float* __restrict__ Y, float* __restrict__ att,
            const float* __restrict__ aw, const float* __restrict__ alphas,
            const float* __restrict__ att0s, int d, int twoHd, float invDT) {
    __shared__ unsigned short As[64 * 32];
    __shared__ unsigned short Bs[64 * 32];
    int z = blockIdx.z, n = z / 3, h = z - n * 3;
    const float* Q  = Y + (i64)n * twoHd * 3000 + (i64)h * d * 3000;
    const float* Kp = Y + (i64)n * twoHd * 3000 + (i64)(3 + h) * d * 3000;
    int Kdim = d * 20;
    int tM = blockIdx.y * 64, tN = blockIdx.x * 64;
    int tid = threadIdx.x, lane = tid & 31, wave = tid >> 5;
    int wm = wave >> 1, wn = wave & 1;
    Acc4 acc; acc_zero(acc);

    for (int k0 = 0; k0 < Kdim; k0 += 32) {
#pragma unroll
        for (int e = tid; e < 2048; e += 128) {
            int r = e >> 5, c = e & 31;
            int kg = k0 + c;
            int u = tM + r;
            As[e] = f2bf(u < 150 ? Q[(i64)kg * 150 + u] : 0.f);
            int v = tN + r;
            Bs[e] = f2bf(v < 150 ? Kp[(i64)kg * 150 + v] : 0.f);
        }
        __syncthreads();
        wmma_step(As, Bs, wm * 32, wn * 32, lane, acc);
        __syncthreads();
    }

    float al = alphas[h];
    int nLoc = lane & 15, mOff = (lane >> 4) * 8;
    float* attz = att + (i64)z * 22500;
    const float* a0 = att0s + (i64)h * 22500;
#pragma unroll
    for (int i = 0; i < 2; ++i)
#pragma unroll
        for (int j = 0; j < 2; ++j) {
            int vg = tN + wn * 32 + j * 16 + nLoc;
            if (vg < 150) {
#pragma unroll
                for (int r = 0; r < 8; ++r) {
                    int ug = tM + wm * 32 + i * 16 + mOff + r;
                    if (ug < 150) {
                        float s = tanhf(acc.a[i][j][r] * invDT) * al;
                        attz[ug * 150 + vg] = s * aw[ug * 150 + vg] + a0[ug * 150 + vg];
                    }
                }
            }
        }
}

// ---------------------------------------------------------------------------
// Apply attention: XS[n, h*C+c, t, v] = sum_u X[n,c,t,u] * att[n,h,u,v]
// A = X[n] as (C*20) x 150 row-major; B = att[n,h] 150x150. M multiple of 64.
// ---------------------------------------------------------------------------
__global__ __launch_bounds__(128)
void k_gemm_xs(const float* __restrict__ X, const float* __restrict__ att,
               float* __restrict__ XS, int C) {
    __shared__ unsigned short As[64 * 32];
    __shared__ unsigned short Bs[64 * 32];
    int z = blockIdx.z, n = z / 3, h = z - n * 3;
    const float* A = X + (i64)n * C * 3000;
    const float* B = att + (i64)z * 22500;
    int tM = blockIdx.y * 64, tN = blockIdx.x * 64;
    int tid = threadIdx.x, lane = tid & 31, wave = tid >> 5;
    int wm = wave >> 1, wn = wave & 1;
    Acc4 acc; acc_zero(acc);

    for (int k0 = 0; k0 < 160; k0 += 32) {
#pragma unroll
        for (int e = tid; e < 2048; e += 128) {
            int r = e >> 5, c = e & 31;
            int kg = k0 + c;
            As[e] = f2bf(kg < 150 ? A[(i64)(tM + r) * 150 + kg] : 0.f);
            int v = tN + r;
            Bs[e] = f2bf((kg < 150 && v < 150) ? B[(i64)kg * 150 + v] : 0.f);
        }
        __syncthreads();
        wmma_step(As, Bs, wm * 32, wn * 32, lane, acc);
        __syncthreads();
    }

    float* outp = XS + (i64)n * 3 * C * 3000 + (i64)h * C * 3000;
    int nLoc = lane & 15, mOff = (lane >> 4) * 8;
#pragma unroll
    for (int i = 0; i < 2; ++i)
#pragma unroll
        for (int j = 0; j < 2; ++j) {
            int vg = tN + wn * 32 + j * 16 + nLoc;
            if (vg < 150) {
#pragma unroll
                for (int r = 0; r < 8; ++r) {
                    int mg = tM + wm * 32 + i * 16 + mOff + r;
                    outp[(i64)mg * 150 + vg] = acc.a[i][j][r];
                }
            }
        }
}

// ---------------------------------------------------------------------------
// Small elementwise kernels
// ---------------------------------------------------------------------------
__global__ void k_reorder(const float* __restrict__ xin, float* __restrict__ X0) {
    int idx = blockIdx.x * 256 + threadIdx.x;
    if (idx >= 32 * 3 * 3000) return;
    int p = idx % 3000; int rest = idx / 3000; int ch = rest % 3; int nm = rest / 3;
    int b = nm >> 1, mp = nm & 1;
    int f = p / 25, j = p - f * 25;
    X0[idx] = xin[((((i64)b * 3 + ch) * 120 + f) * 25 + j) * 2 + mp];
}

__global__ void k_pe(float* __restrict__ pe, int C) {
    int idx = blockIdx.x * 256 + threadIdx.x;
    if (idx >= C * 150) return;
    int v = idx % 150, ch = idx / 150;
    int i = ch >> 1;
    float div = expf(-(float)(2 * i) * 9.210340372f / (float)C);
    float ang = (float)v * div;
    pe[idx] = (ch & 1) ? cosf(ang) : sinf(ang);
}

__global__ void k_aw(const float* __restrict__ spw, const float* __restrict__ bmat,
                     float* __restrict__ aw) {
    int idx = blockIdx.x * 256 + threadIdx.x;
    if (idx < 22500) aw[idx] = spw[idx] * (1.f / 11.f) + bmat[idx];
}

__global__ void k_cvt_bf16(const float* __restrict__ src, unsigned short* __restrict__ dst, int n) {
    int idx = blockIdx.x * 256 + threadIdx.x;
    if (idx < n) dst[idx] = f2bf(src[idx]);
}

__global__ void k_mean(const float* __restrict__ Xf, float* __restrict__ xm, int C) {
    int idx = blockIdx.x * 256 + threadIdx.x;
    if (idx >= 16 * C * 120) return;
    int f = idx % 120; int rest = idx / 120; int ch = rest % C; int b = rest / C;
    float s = 0.f;
    for (int mp = 0; mp < 2; ++mp) {
        const float* p = Xf + ((i64)(b * 2 + mp) * C + ch) * 3000 + f * 25;
        for (int j = 0; j < 25; ++j) s += p[j];
    }
    xm[idx] = s * (1.f / 50.f);
}

// ---------------------------------------------------------------------------
// Classifier head: per (b,c) workgroup does P = Kmat[c] @ x1[b]  (256x119, K=256)
// via WMMA (128x64 block tile, 8 waves, 2x2 passes), then
// out[b,c] = -mean_t sqrt( sum_i (P[i,t]-x2[b,i,t])^2 )
// A operand (Kmat, pre-converted to bf16) is staged with async global->LDS
// copies tracked by ASYNCcnt; B staged with ds stores (needs f32->bf16 cvt).
// ---------------------------------------------------------------------------
__global__ __launch_bounds__(256)
void k_head(const unsigned short* __restrict__ Kbf, const float* __restrict__ xm,
            float* __restrict__ outp) {
    __shared__ __align__(16) unsigned short As[128 * 32];
    __shared__ __align__(16) unsigned short Bs[64 * 32];
    __shared__ float sacc[128];
    __shared__ float rbuf[256];
    int bc = blockIdx.x; int b = bc / 60; int c = bc - b * 60;
    int tid = threadIdx.x, lane = tid & 31, wave = tid >> 5;
    int wm = wave >> 1, wn = wave & 1;                  // 8 waves: 4 (M) x 2 (N), 32x32 each
    for (int i = tid; i < 128; i += 256) sacc[i] = 0.f;
    __syncthreads();
    const unsigned short* A = Kbf + (i64)c * 256 * 256; // [i][j] bf16
    const float* Xb = xm + (i64)b * 256 * 120;          // [j][t]

    for (int mBlk = 0; mBlk < 2; ++mBlk) {
        for (int nBlk = 0; nBlk < 2; ++nBlk) {
            int tM = mBlk * 128, tN = nBlk * 64;
            Acc4 acc; acc_zero(acc);
            for (int k0 = 0; k0 < 256; k0 += 32) {
                // A tile 128x32 bf16: 512 x 16B async copies, 2 per thread.
#pragma unroll
                for (int s = 0; s < 2; ++s) {
                    int seg = tid + s * 256;          // 0..511
                    int r = seg >> 2, q = seg & 3;    // row, 16B quarter
                    const unsigned short* g = A + (i64)(tM + r) * 256 + (k0 + q * 8);
                    unsigned lds = (unsigned)(uintptr_t)(&As[r * 32 + q * 8]);
                    asm volatile("global_load_async_to_lds_b128 %0, %1, off"
                                 :: "v"(lds), "v"(g) : "memory");
                }
                // B tile (transposed) with f32->bf16 conversion.
#pragma unroll
                for (int e = tid; e < 64 * 32; e += 256) {
                    int nl = e >> 5, kk = e & 31;
                    int tg = tN + nl;
                    Bs[e] = f2bf(tg < 119 ? Xb[(i64)(k0 + kk) * 120 + tg] : 0.f);
                }
                asm volatile("s_wait_asynccnt 0x0" ::: "memory");
                __syncthreads();
                wmma_step(As, Bs, wm * 32, wn * 32, lane, acc);
                __syncthreads();
            }
            int nLoc = lane & 15, mOff = (lane >> 4) * 8;
#pragma unroll
            for (int j = 0; j < 2; ++j) {
                int tg = tN + wn * 32 + j * 16 + nLoc;
                if (tg < 119) {
                    float part = 0.f;
#pragma unroll
                    for (int i = 0; i < 2; ++i)
#pragma unroll
                        for (int r = 0; r < 8; ++r) {
                            int ig = tM + wm * 32 + i * 16 + mOff + r;
                            float diff = acc.a[i][j][r] - Xb[(i64)ig * 120 + (tg + 1)];
                            part += diff * diff;
                        }
                    atomicAdd(&sacc[tg], part);
                }
            }
        }
    }
    __syncthreads();
    float ps = 0.f;
    for (int t = tid; t < 119; t += 256) ps += sqrtf(sacc[t]);
    rbuf[tid] = ps;
    __syncthreads();
    for (int s = 128; s > 0; s >>= 1) {
        if (tid < s) rbuf[tid] += rbuf[tid + s];
        __syncthreads();
    }
    if (tid == 0) outp[bc] = -rbuf[0] * (1.f / 119.f);
}

// ---------------------------------------------------------------------------
// Host side
// ---------------------------------------------------------------------------
static inline void conv_launch(hipStream_t st, const float* X, const float* W, const float* bias,
                               const float* bnS, const float* bnT, const float* res,
                               const float* pe, float* out,
                               int M, int K, int Cin, int KH, int KW, int padH, int padW,
                               int lrelu) {
    dim3 g((3000 + 63) / 64, (M + 63) / 64, 32);
    i64 xS = (i64)Cin * 3000, oS = (i64)M * 3000, rS = (i64)M * 3000;
    if (KH == 1 && KW == 1)
        k_gemm_conv<1, 1><<<g, 128, 0, st>>>(X, W, bias, bnS, bnT, res, pe, out, M, K, padH, padW, xS, oS, rS, lrelu);
    else if (KH == 1 && KW == 5)
        k_gemm_conv<1, 5><<<g, 128, 0, st>>>(X, W, bias, bnS, bnT, res, pe, out, M, K, padH, padW, xS, oS, rS, lrelu);
    else
        k_gemm_conv<3, 1><<<g, 128, 0, st>>>(X, W, bias, bnS, bnT, res, pe, out, M, K, padH, padW, xS, oS, rS, lrelu);
}

enum { QKV_W, QKV_B, ALPHAS, ATT0S, OUT_W, OUT_B, FF_W, FF_B, TCN_W, TCN_B,
       OUT_BN_S, OUT_BN_T, FF_BN_S, FF_BN_T, TCN_BN_S, TCN_BN_T,
       RESS_W, RESS_B, RESS_BN_S, RESS_BN_T, REST_W, REST_B, REST_BN_S, REST_BN_T };

extern "C" void kernel_launch(void* const* d_in, const int* in_sizes, int n_in,
                              void* d_out, int out_size, void* d_ws, size_t ws_size,
                              hipStream_t stream) {
    const float* x_in    = (const float*)d_in[0];
    const float* spw     = (const float*)d_in[1];
    const float* input_w = (const float*)d_in[2];
    const float* input_b = (const float*)d_in[3];
    const float* bmat    = (const float*)d_in[4];
    const float* Kmat    = (const float*)d_in[5];
    const float* P[3][24];
    int idx = 6;
    for (int k = 0; k < 3; ++k) {
        for (int j = 0; j < 16; ++j) P[k][j] = (const float*)d_in[idx++];
        if (k > 0) { for (int j = 16; j < 24; ++j) P[k][j] = (const float*)d_in[idx++]; }
        else       { for (int j = 16; j < 24; ++j) P[k][j] = (const float*)0; }
    }
    const float* input_bn_s = (const float*)d_in[idx++];
    const float* input_bn_t = (const float*)d_in[idx++];

    // workspace carve-up
    char* wsb = (char*)d_ws;
    size_t off = 0;
    auto F = [&](size_t n) { float* p = (float*)(wsb + off); off += ((n * 4 + 255) / 256) * 256; return p; };
    float* pe64  = F(64 * 150);
    float* pe128 = F(128 * 150);
    float* awb   = F(22500);
    float* B0    = F((size_t)32 * 256 * 3000);
    float* B1    = F((size_t)32 * 256 * 3000);
    float* B2    = F((size_t)32 * 256 * 3000);
    float* B3    = F((size_t)32 * 256 * 3000);
    float* Ybuf  = F((size_t)32 * 384 * 3000);   // qkv output, later reused for XS
    float* attb  = F((size_t)96 * 22500);
    float* xmb   = F((size_t)16 * 256 * 120);
    unsigned short* kbf = (unsigned short*)F((size_t)60 * 256 * 256 / 2 + 64); // bf16 Kmat

    k_pe<<<(64 * 150 + 255) / 256, 256, 0, stream>>>(pe64, 64);
    k_pe<<<(128 * 150 + 255) / 256, 256, 0, stream>>>(pe128, 128);
    k_aw<<<(22500 + 255) / 256, 256, 0, stream>>>(spw, bmat, awb);
    k_reorder<<<(288000 + 255) / 256, 256, 0, stream>>>(x_in, B1);
    k_cvt_bf16<<<(60 * 256 * 256 + 255) / 256, 256, 0, stream>>>(Kmat, kbf, 60 * 256 * 256);

    // input conv: 3 -> 64, BN, lrelu
    conv_launch(stream, B1, input_w, input_b, input_bn_s, input_bn_t,
                (const float*)0, (const float*)0, B0, 64, 3, 3, 1, 1, 0, 0, 1);

    const int CIN[3]  = {64, 64, 128};
    const int COUT[3] = {64, 128, 256};
    const int DD[3]   = {16, 32, 64};
    float* XIN[3] = {B0, B3, B1};
    float* XR[3]  = {B0, B0, B0};   // blk0: identity residual = input
    float* OC[3]  = {B1, B1, B3};
    float* FFO[3] = {B2, B2, B2};
    float* XRT[3] = {B2, B0, B0};   // blk0: xrt = ff output
    float* TCO[3] = {B3, B1, B3};

    for (int k = 0; k < 3; ++k) {
        int cin = CIN[k], cout = COUT[k], d = DD[k];
        const float* pe = (cin == 64) ? pe64 : pe128;
        const float** p = P[k];

        // qkv (1x1, + positional encoding on input)
        conv_launch(stream, XIN[k], p[QKV_W], p[QKV_B], 0, 0, 0, pe, Ybuf,
                    6 * d, cin, cin, 1, 1, 0, 0, 0);

        // attention scores
        dim3 ga(3, 3, 96);
        k_attn<<<ga, 128, 0, stream>>>(Ybuf, attb, awb, p[ALPHAS], p[ATT0S],
                                       d, 6 * d, 1.f / (float)(d * 20));

        // apply attention (XS overwrites Ybuf; qkv output is dead)
        dim3 gx(3, (cin * 20) / 64, 96);
        k_gemm_xs<<<gx, 128, 0, stream>>>(XIN[k], attb, Ybuf, cin);

        // residual branch
        if (k > 0)
            conv_launch(stream, XIN[k], p[RESS_W], p[RESS_B], p[RESS_BN_S], p[RESS_BN_T],
                        0, 0, XR[k], cout, cin, cin, 1, 1, 0, 0, 0);

        // out conv: (cout, 3*cin, 1, 5), pad W=2, BN, +xr, lrelu
        conv_launch(stream, Ybuf, p[OUT_W], p[OUT_B], p[OUT_BN_S], p[OUT_BN_T],
                    XR[k], 0, OC[k], cout, 3 * cin * 5, 3 * cin, 1, 5, 0, 2, 1);

        // ff 1x1, BN, +xr, lrelu
        conv_launch(stream, OC[k], p[FF_W], p[FF_B], p[FF_BN_S], p[FF_BN_T],
                    XR[k], 0, FFO[k], cout, cout, cout, 1, 1, 0, 0, 1);

        // temporal residual branch
        if (k > 0)
            conv_launch(stream, FFO[k], p[REST_W], p[REST_B], p[REST_BN_S], p[REST_BN_T],
                        0, 0, XRT[k], cout, cout, cout, 1, 1, 0, 0, 0);

        // tcn: (cout, cout, 3, 1), pad H=1, BN, +xrt, lrelu
        conv_launch(stream, FFO[k], p[TCN_W], p[TCN_B], p[TCN_BN_S], p[TCN_BN_T],
                    XRT[k], 0, TCO[k], cout, cout * 3, cout, 3, 1, 1, 0, 1);
    }

    // mean over persons and joints -> xm[16][256][120]
    k_mean<<<(16 * 256 * 120 + 255) / 256, 256, 0, stream>>>(B3, xmb, 256);

    // classifier head -> out[16][60]
    k_head<<<960, 256, 0, stream>>>(kbf, xmb, (float*)d_out);
}
